// MultiHeadAttention_43404939493624
// MI455X (gfx1250) — matrix operand order repaired and checked by reading, not compiled, not run on hardware
//
#include <hip/hip_runtime.h>
#include <hip/hip_bf16.h>

// ---------------------------------------------------------------------------
// Problem constants (match reference)
// ---------------------------------------------------------------------------
constexpr int BB = 4;
constexpr int TT = 2048;
constexpr int CC = 2048;
constexpr int HH = 16;
constexpr int HS = 128;           // head size
constexpr int M_ROWS = BB * TT;   // 8192

typedef _Float16 v16h __attribute__((ext_vector_type(16)));
typedef float    v8f  __attribute__((ext_vector_type(8)));

// ---------------------------------------------------------------------------
// CDNA5 async global->LDS staging (ASYNCcnt path). Guarded so compilation
// cannot regress if the builtin is absent on this toolchain.
// The builtin's params are typed: (AS1 int4*, AS3 int4*, imm, imm) where int4
// is the *generic-vector* (vector_size) flavor — match it exactly.
// ---------------------------------------------------------------------------
#if defined(__has_builtin)
#if __has_builtin(__builtin_amdgcn_global_load_async_to_lds_b128)
#define HAVE_ASYNC_LDS 1
#endif
#endif
#ifndef HAVE_ASYNC_LDS
#define HAVE_ASYNC_LDS 0
#endif

#if HAVE_ASYNC_LDS
typedef int i32x4_ __attribute__((vector_size(16)));
// imm offset is applied to BOTH the global and LDS address (ISA 08 §4.4).
#define ASYNC_CP_B128(g, l, imm)                                             \
  __builtin_amdgcn_global_load_async_to_lds_b128(                            \
      (__attribute__((address_space(1))) i32x4_*)(g),                        \
      (__attribute__((address_space(3))) i32x4_*)(l), (imm), 0)
__device__ __forceinline__ void async_wait_all() {
  asm volatile("s_wait_asynccnt 0x0" ::: "memory");
}
#else
#define ASYNC_CP_B128(g, l, imm)                                             \
  do {                                                                       \
    *(uint4*)((char*)(l) + (imm)) =                                          \
        *(const uint4*)((const char*)(g) + (imm));                           \
  } while (0)
__device__ __forceinline__ void async_wait_all() {}
#endif

// ---------------------------------------------------------------------------
// WMMA helpers
// ---------------------------------------------------------------------------
__device__ __forceinline__ v8f zero8() {
  v8f z;
#pragma unroll
  for (int i = 0; i < 8; ++i) z[i] = 0.0f;
  return z;
}

__device__ __forceinline__ v8f wmma_f16(v16h a, v16h b, v8f c) {
  // D = A(16x32 f16) * B(32x16 f16) + C(16x16 f32)
  return __builtin_amdgcn_wmma_f32_16x16x32_f16(
      /*neg_a=*/false, a, /*neg_b=*/false, b,
      /*c_mod=*/(short)0, c, /*reuse_a=*/false, /*reuse_b=*/false);
}

// A-fragment (16x32, f16), row-major source, leading dim `ld` (elements).
// Wave32 layout: lane -> (m = lane&15, kh = lane>>4); half j ->
//   k = (j>>3)*16 + kh*8 + (j&7)  => two contiguous 8-half chunks.
__device__ __forceinline__ v16h load_a_frag(const _Float16* __restrict__ base,
                                            int ld) {
  const int lane = threadIdx.x & 31;
  const int m = lane & 15;
  const int kh = lane >> 4;
  const _Float16* p = base + (size_t)m * ld + kh * 8;
  v16h a;
#pragma unroll
  for (int j = 0; j < 8; ++j) {
    a[j]     = p[j];
    a[j + 8] = p[16 + j];
  }
  return a;
}

// B-fragment (32x16, f16) from a source stored as B^T: row n holds the K dim
// contiguously with leading dim `ld`. lane -> (n = lane&15, kh = lane>>4);
// half j -> k = (j>>3)*16 + kh*8 + (j&7)  => two contiguous 8-half chunks.
__device__ __forceinline__ v16h load_b_frag(const _Float16* __restrict__ base,
                                            int ld) {
  const int lane = threadIdx.x & 31;
  const int n = lane & 15;
  const int kh = lane >> 4;
  const _Float16* p = base + (size_t)n * ld + kh * 8;
  v16h b;
#pragma unroll
  for (int j = 0; j < 8; ++j) {
    b[j]     = p[j];
    b[j + 8] = p[16 + j];
  }
  return b;
}

__device__ __forceinline__ float rmax16(float v) {
  v = fmaxf(v, __shfl_xor(v, 1, 16));
  v = fmaxf(v, __shfl_xor(v, 2, 16));
  v = fmaxf(v, __shfl_xor(v, 4, 16));
  v = fmaxf(v, __shfl_xor(v, 8, 16));
  return v;
}

__device__ __forceinline__ float rsum16(float v) {
  v += __shfl_xor(v, 1, 16);
  v += __shfl_xor(v, 2, 16);
  v += __shfl_xor(v, 4, 16);
  v += __shfl_xor(v, 8, 16);
  return v;
}

// ---------------------------------------------------------------------------
// f32 -> f16 conversion (grid-stride)
// ---------------------------------------------------------------------------
__global__ void cvt_f32_to_f16(const float* __restrict__ in,
                               _Float16* __restrict__ out, long n) {
  long i = (long)blockIdx.x * blockDim.x + threadIdx.x;
  const long stride = (long)gridDim.x * blockDim.x;
  for (; i < n; i += stride) out[i] = (_Float16)in[i];
}

// ---------------------------------------------------------------------------
// GEMM: D[M,N] = A[M,K] * W[N,K]^T  (nn.Linear semantics)
// Workgroup tile 256x128, 8 waves of 64x64 (16 WMMA / 16 ds_load_b128 per
// K-chunk). K staged in 32-wide chunks, DOUBLE-BUFFERED via async global->LDS
// copies: issue chunk k+1 right after the barrier, compute chunk k, wait
// ASYNCcnt only at the next iteration top.
// WITH_BIAS=false -> f16 output; WITH_BIAS=true -> f32 output + bias.
// M multiple of 256; N multiple of 128; K multiple of 32.
// ---------------------------------------------------------------------------
template <bool WITH_BIAS>
__global__ __launch_bounds__(256) void gemm_f16_wmma(
    const _Float16* __restrict__ A, const _Float16* __restrict__ W,
    _Float16* __restrict__ outh, float* __restrict__ outf,
    const float* __restrict__ bias, int M, int N, int K) {
  __shared__ __align__(16) _Float16 As[2][256][40];  // 40 KiB (32k + pad)
  __shared__ __align__(16) _Float16 Ws[2][128][40];  // 20 KiB

  const int tid  = threadIdx.x;
  const int wave = tid >> 5;
  const int lane = tid & 31;
  const int mbase = blockIdx.y * 256;
  const int nbase = blockIdx.x * 128;
  const int wm = (wave >> 1) * 64;  // wave row offset: 0/64/128/192
  const int wn = (wave & 1) * 64;   // wave col offset: 0/64

  v8f acc[4][4];
#pragma unroll
  for (int i = 0; i < 4; ++i)
#pragma unroll
    for (int j = 0; j < 4; ++j) acc[i][j] = zero8();

  // Staging assignment: A -> one 64B row-chunk per thread; W -> half row.
  const int arow = tid;               // 0..255
  const int wrow = tid >> 1;          // 0..127
  const int wofs = (tid & 1) * 16;    // halves (0 or 16)
  const _Float16* gA = A + (size_t)(mbase + arow) * K;
  const _Float16* gW = W + (size_t)(nbase + wrow) * K + wofs;

  const int nk = K >> 5;

  // Preload chunk 0 into buffer 0.
  ASYNC_CP_B128(gA, &As[0][arow][0], 0);
  ASYNC_CP_B128(gA, &As[0][arow][0], 16);
  ASYNC_CP_B128(gA, &As[0][arow][0], 32);
  ASYNC_CP_B128(gA, &As[0][arow][0], 48);
  ASYNC_CP_B128(gW, &Ws[0][wrow][wofs], 0);
  ASYNC_CP_B128(gW, &Ws[0][wrow][wofs], 16);

  for (int ik = 0; ik < nk; ++ik) {
    const int cur = ik & 1;
    async_wait_all();    // my chunk-`ik` copies have landed in LDS
    __syncthreads();     // everyone's copies landed; prev-iter readers done
    if (ik + 1 < nk) {   // kick off chunk ik+1 into the other buffer
      const _Float16* pa = gA + (ik + 1) * 32;
      const _Float16* pw = gW + (ik + 1) * 32;
      ASYNC_CP_B128(pa, &As[1 - cur][arow][0], 0);
      ASYNC_CP_B128(pa, &As[1 - cur][arow][0], 16);
      ASYNC_CP_B128(pa, &As[1 - cur][arow][0], 32);
      ASYNC_CP_B128(pa, &As[1 - cur][arow][0], 48);
      ASYNC_CP_B128(pw, &Ws[1 - cur][wrow][wofs], 0);
      ASYNC_CP_B128(pw, &Ws[1 - cur][wrow][wofs], 16);
    }

    v16h af[4];
#pragma unroll
    for (int i = 0; i < 4; ++i)
      af[i] = load_a_frag(&As[cur][wm + i * 16][0], 40);
#pragma unroll
    for (int j = 0; j < 4; ++j) {
      const v16h bf = load_b_frag(&Ws[cur][wn + j * 16][0], 40);
#pragma unroll
      for (int i = 0; i < 4; ++i) acc[i][j] = wmma_f16(af[i], bf, acc[i][j]);
    }
  }

  // Epilogue. C/D layout: element r of v8f = (row r + 8*(lane>>4), col lane&15)
  const int n_in = lane & 15;
  const int mh = lane >> 4;
#pragma unroll
  for (int mt = 0; mt < 4; ++mt) {
#pragma unroll
    for (int nt = 0; nt < 4; ++nt) {
      const int col = nbase + wn + nt * 16 + n_in;
#pragma unroll
      for (int r = 0; r < 8; ++r) {
        const int row = mbase + wm + mt * 16 + r + 8 * mh;
        const float v = acc[mt][nt][r];
        if (WITH_BIAS) {
          outf[(size_t)row * N + col] = v + bias[col];
        } else {
          outh[(size_t)row * N + col] = (_Float16)v;
        }
      }
    }
  }
}

// ---------------------------------------------------------------------------
// Flash attention (causal). qkv: f16 [B*T, 3C]; out: f16 [B*T, C].
// One workgroup (4 waves) per 64 query rows of one (b,h). Each wave owns 16
// query rows, holding O (16x128 f32) in 8 WMMA accumulators. Streams 32-key
// blocks: K staged via async global->LDS, V transposed into LDS, S via WMMA,
// online softmax with half-wave shuffles, P re-laid out through LDS,
// O += P*V via WMMA.
// ---------------------------------------------------------------------------
__global__ __launch_bounds__(128) void flash_attn_wmma(
    const _Float16* __restrict__ qkv, _Float16* __restrict__ out) {
  __shared__ __align__(16) _Float16 Ks[32][136];    // [key][d]
  __shared__ __align__(16) _Float16 Vt[128][40];    // [d][key] (transposed)
  __shared__ __align__(16) _Float16 Ps[4][16][40];  // per-wave P tile [m][k]

  const int tid  = threadIdx.x;
  const int wave = tid >> 5;
  const int lane = tid & 31;
  const int n_in = lane & 15;
  const int mh   = lane >> 4;

  const int bid  = blockIdx.x;
  const int qblk = bid & 31;          // T/64 = 32 query blocks
  const int h    = (bid >> 5) & 15;
  const int b    = bid >> 9;
  const int qbase = qblk * 64;
  const int qrow0 = qbase + wave * 16;  // this wave's first query row
  const size_t rs = (size_t)3 * CC;     // qkv row stride

  // Load Q fragments (16 rows x 128 d) once: 4 A-frags along the head dim.
  const _Float16* qptr = qkv + (size_t)(b * TT + qrow0) * rs + h * HS;
  v16h qf[4];
#pragma unroll
  for (int ks = 0; ks < 4; ++ks) qf[ks] = load_a_frag(qptr + ks * 32, (int)rs);

  v8f o[8];
#pragma unroll
  for (int t = 0; t < 8; ++t) o[t] = zero8();
  float mrow[8], lrow[8];
#pragma unroll
  for (int r = 0; r < 8; ++r) { mrow[r] = -1.0e30f; lrow[r] = 0.0f; }

  const float scale = 0.08838834764831845f;  // 1/sqrt(128)
  const int nblk = (qbase + 64) >> 5;        // key blocks needed (causal)

  for (int jb = 0; jb < nblk; ++jb) {
    const int kbase = jb * 32;
    __syncthreads();  // prior iteration done with Ks/Vt
    {
      // Async K staging: 32 keys x 128 halves, [key][d]; 64B per thread.
      const int r = tid >> 2;
      const int off = (tid & 3) * 32;
      const _Float16* gk =
          qkv + (size_t)(b * TT + kbase + r) * rs + CC + h * HS + off;
      ASYNC_CP_B128(gk, &Ks[r][off], 0);
      ASYNC_CP_B128(gk, &Ks[r][off], 16);
      ASYNC_CP_B128(gk, &Ks[r][off], 32);
      ASYNC_CP_B128(gk, &Ks[r][off], 48);
      // Cooperative V load, transposed into Vt[d][key].
      const _Float16* gv =
          qkv + (size_t)(b * TT + kbase + r) * rs + 2 * CC + h * HS + off;
#pragma unroll
      for (int j = 0; j < 32; ++j) Vt[off + j][r] = gv[j];
    }
    async_wait_all();
    __syncthreads();

    // S = Q * K^T : two 16x16 tiles covering 32 keys.
    v8f s0 = zero8(), s1 = zero8();
#pragma unroll
    for (int ks = 0; ks < 4; ++ks) {
      s0 = wmma_f16(qf[ks], load_b_frag(&Ks[0][ks * 32], 136), s0);
      s1 = wmma_f16(qf[ks], load_b_frag(&Ks[16][ks * 32], 136), s1);
    }

    // Scale + causal mask (finite sentinel avoids NaN in streaming softmax).
    const int kp0 = kbase + n_in;
    const int kp1 = kbase + 16 + n_in;
#pragma unroll
    for (int r = 0; r < 8; ++r) {
      const int qp = qrow0 + r + 8 * mh;
      s0[r] = (kp0 <= qp) ? s0[r] * scale : -1.0e30f;
      s1[r] = (kp1 <= qp) ? s1[r] * scale : -1.0e30f;
    }

    // Online softmax update.
    float alpha[8];
#pragma unroll
    for (int r = 0; r < 8; ++r) {
      const float rm = rmax16(fmaxf(s0[r], s1[r]));
      const float mn = fmaxf(mrow[r], rm);
      alpha[r] = __expf(mrow[r] - mn);
      mrow[r] = mn;
      s0[r] = __expf(s0[r] - mn);
      s1[r] = __expf(s1[r] - mn);
      lrow[r] = lrow[r] * alpha[r] + rsum16(s0[r] + s1[r]);
    }
#pragma unroll
    for (int t = 0; t < 8; ++t)
#pragma unroll
      for (int r = 0; r < 8; ++r) o[t][r] *= alpha[r];

    // Re-layout P (f32 C-tile -> f16 A-frag) through LDS.
#pragma unroll
    for (int r = 0; r < 8; ++r) {
      const int m = r + 8 * mh;
      Ps[wave][m][n_in]      = (_Float16)s0[r];
      Ps[wave][m][16 + n_in] = (_Float16)s1[r];
    }
    __syncthreads();  // uniform across waves; makes P (and Vt) safely ordered

    const v16h pf = load_a_frag(&Ps[wave][0][0], 40);
#pragma unroll
    for (int t = 0; t < 8; ++t)
      o[t] = wmma_f16(pf, load_b_frag(&Vt[t * 16][0], 40), o[t]);
  }

  // Normalize and store: out[b*T + q][h*HS + d], f16.
  float inv[8];
#pragma unroll
  for (int r = 0; r < 8; ++r) inv[r] = 1.0f / lrow[r];
#pragma unroll
  for (int t = 0; t < 8; ++t) {
#pragma unroll
    for (int r = 0; r < 8; ++r) {
      const int m = r + 8 * mh;
      out[(size_t)(b * TT + qrow0 + m) * CC + h * HS + t * 16 + n_in] =
          (_Float16)(o[t][r] * inv[r]);
    }
  }
}

// ---------------------------------------------------------------------------
// Launch
// ---------------------------------------------------------------------------
extern "C" void kernel_launch(void* const* d_in, const int* in_sizes, int n_in,
                              void* d_out, int out_size, void* d_ws,
                              size_t ws_size, hipStream_t stream) {
  const float* x      = (const float*)d_in[0];  // [B,T,C]
  const float* w_qkv  = (const float*)d_in[1];  // [3C,C]
  const float* w_proj = (const float*)d_in[2];  // [C,C]
  const float* b_proj = (const float*)d_in[3];  // [C]
  float* out = (float*)d_out;                   // [B,T,C]

  char* ws = (char*)d_ws;
  _Float16* xh     = (_Float16*)(ws);                         // 32 MiB
  _Float16* wqkvh  = (_Float16*)(ws + (size_t)33554432);      // 24 MiB
  _Float16* wprojh = (_Float16*)(ws + (size_t)58720256);      //  8 MiB
  _Float16* qkvh   = (_Float16*)(ws + (size_t)67108864);      // 96 MiB
  _Float16* attnh  = (_Float16*)(ws + (size_t)167772160);     // 32 MiB

  // 1) f32 -> f16 conversions
  cvt_f32_to_f16<<<4096, 256, 0, stream>>>(x, xh, (long)M_ROWS * CC);
  cvt_f32_to_f16<<<4096, 256, 0, stream>>>(w_qkv, wqkvh, (long)3 * CC * CC);
  cvt_f32_to_f16<<<2048, 256, 0, stream>>>(w_proj, wprojh, (long)CC * CC);

  // 2) QKV = x @ w_qkv^T   (M=8192, N=6144, K=2048) -> f16
  gemm_f16_wmma<false><<<dim3(3 * CC / 128, M_ROWS / 256), 256, 0, stream>>>(
      xh, wqkvh, qkvh, nullptr, nullptr, M_ROWS, 3 * CC, CC);

  // 3) Causal flash attention -> f16 [B*T, C]
  flash_attn_wmma<<<BB * HH * (TT / 64), 128, 0, stream>>>(qkvh, attnh);

  // 4) out = attn @ w_proj^T + b_proj  (M=8192, N=2048, K=2048) -> f32
  gemm_f16_wmma<true><<<dim3(CC / 128, M_ROWS / 256), 256, 0, stream>>>(
      attnh, wprojh, nullptr, out, b_proj, M_ROWS, CC, CC);
}
